// Net_5428838662222
// MI455X (gfx1250) — matrix operand order repaired and checked by reading, not compiled, not run on hardware
//
#include <hip/hip_runtime.h>

#define DEV __device__ __forceinline__

typedef _Float16 h16;
typedef __attribute__((ext_vector_type(8)))  _Float16 v8h;
typedef __attribute__((ext_vector_type(16))) _Float16 v16h;
typedef __attribute__((ext_vector_type(8)))  float    v8f;
typedef __attribute__((ext_vector_type(4)))  unsigned int v4u;
typedef __attribute__((ext_vector_type(8)))  int v8i;
typedef __attribute__((ext_vector_type(4)))  int v4i;

// problem dims
constexpr int Bt  = 16;          // batch elements per workgroup
constexpr int Lp  = 8;           // padded sequence (L=7 -> 8)
constexpr int R   = Bt * Lp;     // 128 token rows per workgroup
constexpr int Hn  = 4;
constexpr int LDA = 72;          // f16 activation leading dim (bank-friendly)
constexpr int LDV = 152;         // V-transposed column count (needs >= 15*8+31+1)
constexpr int LDO = 68;          // fp32 accumulator leading dim
constexpr int ROWQ = 136;        // Q/K rows incl. attention-tile overrun pad
constexpr int LDD = 264;         // decoder concat buffer ld (256 + pad)

// LDS arena offsets (bytes, all 16B aligned)
constexpr int OFF_XA = 0;                         // 128x72 f16   activations
constexpr int OFF_XB = OFF_XA + R * LDA * 2;      // 128x72 f16   scratch
constexpr int OFF_QB = OFF_XB + R * LDA * 2;      // 136x72 f16   Q (enc) / V rowmajor (dec)
constexpr int OFF_KB = OFF_QB + ROWQ * LDA * 2;   // 136x72 f16   K
constexpr int OFF_VT = OFF_KB + ROWQ * LDA * 2;   // 64x152 f16   V transposed
constexpr int OFF_OC = OFF_VT + 64 * LDV * 2;     // 136x68 f32   accumulator
constexpr int OFF_SC = OFF_OC + ROWQ * LDO * 4;   // 16x256 f32   score scratch
constexpr int OFF_AT = OFF_SC + 16 * 256 * 4;     // 16x(16x32) f16 attn tiles / dec out
constexpr int OFF_W0 = OFF_AT + 16 * 512 * 2;     // 64x72 f16 weight (transposed)
constexpr int OFF_W1 = OFF_W0 + 64 * LDA * 2;
constexpr int OFF_W2 = OFF_W1 + 64 * LDA * 2;
constexpr int OFF_OD = OFF_W2 + 64 * LDA * 2;     // 16x264 f16 decoder concat
constexpr int OFF_WS = OFF_OD + 16 * LDD * 2;     // 3x(64x64) f32 TDM staging
constexpr int SMEM   = OFF_WS + 3 * 64 * 64 * 4;  // ~245 KB dynamic LDS

DEV v8f vzero8() { v8f c; \
  _Pragma("unroll") for (int i = 0; i < 8; i++) c[i] = 0.f; return c; }

// A fragment (16x32 f16): lane m<16 holds K {k0..k0+7, k0+16..k0+23}, lane m+16 the +8 shifted set
DEV v16h ldsA(const h16* base, int ld, int row0, int k0) {
  int lane = threadIdx.x & 31;
  int m = lane & 15, hi = lane >> 4;
  const h16* p = base + (row0 + m) * ld + k0 + hi * 8;
  v8h a = *(const v8h*)p;
  v8h b = *(const v8h*)(p + 16);
  v16h r;
#pragma unroll
  for (int i = 0; i < 8; i++) { r[i] = a[i]; r[i + 8] = b[i]; }
  return r;
}

// B fragment (32x16 f16) from column-major (transposed) storage: lane n holds col n, K {k0..k0+7, k0+16..k0+23}, lane n+16 the +8 shifted set
DEV v16h ldsB(const h16* base, int ld, int n0, int k0) {
  int lane = threadIdx.x & 31;
  int n = lane & 15, hi = lane >> 4;
  const h16* p = base + (n0 + n) * ld + k0 + hi * 8;
  v8h a = *(const v8h*)p;
  v8h b = *(const v8h*)(p + 16);
  v16h r;
#pragma unroll
  for (int i = 0; i < 8; i++) { r[i] = a[i]; r[i + 8] = b[i]; }
  return r;
}

DEV v8f wmma_f16(v16h a, v16h b, v8f c) {
  return __builtin_amdgcn_wmma_f32_16x16x32_f16(false, a, false, b, (short)0, c, false, false);
}

// ---- Tensor Data Mover: DMA a kRows x nCols f32 tile (row stride srcld) into LDS (compact) ----
// D# per CDNA5 ISA ch.8: group0 = {flags/count, lds_addr, global_addr lo, global_addr hi | type=2}
// group1 = {mask/data_size, dims, tile dims, strides}; groups 2/3 (and the trailing group on
// this 6-arg clang-23 builtin) are zero for a 2-D tensor.
DEV void tdm_load(unsigned lds_off, const float* src, int srcld, int nCols, int kRows) {
  unsigned long long ga = (unsigned long long)(size_t)src;
  v4u g0;
  g0[0] = 1u;                                   // count=1, user mode, no gather
  g0[1] = lds_off;                              // lds_addr (bytes)
  g0[2] = (unsigned)ga;                         // global_addr[31:0]
  g0[3] = ((unsigned)(ga >> 32) & 0x01FFFFFFu) | 0x80000000u; // addr[56:32] | type=2
  v8i g1;
  g1[0] = 0x20000;                              // workgroup_mask=0, data_size=2 (4B)
  g1[1] = (nCols & 0xFFFF) << 16;               // tensor_dim0 lo16
  g1[2] = ((nCols >> 16) & 0xFFFF) | ((kRows & 0xFFFF) << 16); // dim0 hi | dim1 lo
  g1[3] = ((kRows >> 16) & 0xFFFF) | ((nCols & 0xFFFF) << 16); // dim1 hi | tile_dim0
  g1[4] = kRows & 0xFFFF;                       // tile_dim1 (tile_dim2=0)
  g1[5] = srcld;                                // tensor_dim0_stride lo32
  g1[6] = 0;
  g1[7] = 0;
  v4i g2; g2[0] = 0; g2[1] = 0; g2[2] = 0; g2[3] = 0;
  v4i g3; g3[0] = 0; g3[1] = 0; g3[2] = 0; g3[3] = 0;
  v8i g4;
#pragma unroll
  for (int i = 0; i < 8; i++) g4[i] = 0;
  __builtin_amdgcn_tensor_load_to_lds(g0, g1, g2, g3, g4, 0);
}

// convert staged f32 tile (LDS, compact [kRows][nCols]) -> f16 transposed [n][ldd]
DEV void cvtWT(h16* dst, int ldd, const float* ws, int nCols, int kRows) {
  for (int i = threadIdx.x; i < kRows * nCols; i += 256) {
    int k = i / nCols, n = i - k * nCols;
    dst[n * ldd + k] = (h16)ws[i];
  }
}

__global__ __launch_bounds__(256)
void xformer_mega(const int* __restrict__ tokens, const float* __restrict__ emb,
                  const float* __restrict__ pos, const float* __restrict__ qparam,
                  const float* __restrict__ in_w, const float* __restrict__ in_b,
                  const float* __restrict__ e_wq, const float* __restrict__ e_wk,
                  const float* __restrict__ e_wv, const float* __restrict__ e_wo,
                  const float* __restrict__ e_bo, const float* __restrict__ e_g1,
                  const float* __restrict__ e_b1, const float* __restrict__ e_c1w,
                  const float* __restrict__ e_c1b, const float* __restrict__ e_c2w,
                  const float* __restrict__ e_c2b, const float* __restrict__ e_g2,
                  const float* __restrict__ e_b2, const float* __restrict__ d_wq,
                  const float* __restrict__ d_wk, const float* __restrict__ d_wv,
                  const float* __restrict__ d_wo, const float* __restrict__ d_bo,
                  const float* __restrict__ fc_w, const float* __restrict__ fc_b,
                  float* __restrict__ out) {
  (void)qparam; (void)d_wq; (void)d_wk;  // dead under softmax(axis=q) with q-len 1

  extern __shared__ char smem[];
  h16*   XA = (h16*)(smem + OFF_XA);
  h16*   XB = (h16*)(smem + OFF_XB);
  h16*   QB = (h16*)(smem + OFF_QB);
  h16*   KB = (h16*)(smem + OFF_KB);
  h16*   VT = (h16*)(smem + OFF_VT);
  float* OC = (float*)(smem + OFF_OC);
  float* SC = (float*)(smem + OFF_SC);
  h16*   AT = (h16*)(smem + OFF_AT);
  h16*   W0 = (h16*)(smem + OFF_W0);
  h16*   W1 = (h16*)(smem + OFF_W1);
  h16*   W2 = (h16*)(smem + OFF_W2);
  h16*   ODC = (h16*)(smem + OFF_OD);
  float* WSf = (float*)(smem + OFF_WS);

  const int tid = threadIdx.x;
  const int w = tid >> 5, lane = tid & 31;
  const int hi = lane >> 4, ln = lane & 15;
  const int b0 = blockIdx.x * Bt;

  // zero whole arena once (pad rows/cols of Q/K/V/AT must be 0)
  for (int i = tid; i < SMEM / 4; i += 256) ((unsigned int*)smem)[i] = 0u;
  __syncthreads();

  // ---- S0: TDM-stage in_w[0:64,:]; gather embeddings into XB meanwhile ----
  if (w == 0) tdm_load(OFF_WS, in_w, 64, 64, 64);
  for (int i = tid; i < R * 64; i += 256) {
    int r = i >> 6, d = i & 63, l = r & 7, bb = r >> 3;
    float v = 0.f;
    if (l < 7) { int t = tokens[(b0 + bb) * 7 + l]; v = emb[t * 64 + d]; }
    XB[r * LDA + d] = (h16)v;
  }
  if (w == 0) __builtin_amdgcn_s_wait_tensorcnt(0);
  __syncthreads();
  cvtWT(W0, LDA, WSf, 64, 64);
  __syncthreads();

  // ---- S1: input projection + pos rank-1 + bias -> XA ----
  for (int j = w; j < 32; j += 8) {
    int rt = j >> 2, ct = j & 3;
    v8f c = vzero8();
    c = wmma_f16(ldsA(XB, LDA, rt * 16, 0),  ldsB(W0, LDA, ct * 16, 0),  c);
    c = wmma_f16(ldsA(XB, LDA, rt * 16, 32), ldsB(W0, LDA, ct * 16, 32), c);
    int col = ct * 16 + ln;
    float wrow = in_w[64 * 64 + col], bias = in_b[col];
#pragma unroll
    for (int vg = 0; vg < 8; vg++) {
      int row = rt * 16 + vg + hi * 8, l = row & 7;
      float pv = (l < 7) ? pos[l] : 0.f;
      XA[row * LDA + col] = (h16)(c[vg] + bias + pv * wrow);
    }
  }
  __syncthreads();

  // ---- S2: encoder MHA (per head; OC accumulates Sum_h O_h @ Wo_h) ----
  for (int h = 0; h < Hn; h++) {
    if (w == 0) {
      tdm_load(OFF_WS,          e_wq + h * 4096, 64, 64, 64);
      tdm_load(OFF_WS + 16384,  e_wk + h * 4096, 64, 64, 64);
      tdm_load(OFF_WS + 32768,  e_wv + h * 4096, 64, 64, 64);
      __builtin_amdgcn_s_wait_tensorcnt(0);
    }
    __syncthreads();
    cvtWT(W0, LDA, WSf,        64, 64);
    cvtWT(W1, LDA, WSf + 4096, 64, 64);
    cvtWT(W2, LDA, WSf + 8192, 64, 64);
    __syncthreads();
    // Q -> QB, K -> KB, V -> VT (transposed)
    for (int j = w; j < 96; j += 8) {
      int o = j >> 5, jj = j & 31, rt = jj >> 2, ct = jj & 3;
      const h16* W = (o == 0) ? W0 : ((o == 1) ? W1 : W2);
      v8f c = vzero8();
      c = wmma_f16(ldsA(XA, LDA, rt * 16, 0),  ldsB(W, LDA, ct * 16, 0),  c);
      c = wmma_f16(ldsA(XA, LDA, rt * 16, 32), ldsB(W, LDA, ct * 16, 32), c);
      if (o == 0) {
#pragma unroll
        for (int vg = 0; vg < 8; vg++) QB[(rt * 16 + vg + hi * 8) * LDA + ct * 16 + ln] = (h16)c[vg];
      } else if (o == 1) {
#pragma unroll
        for (int vg = 0; vg < 8; vg++) KB[(rt * 16 + vg + hi * 8) * LDA + ct * 16 + ln] = (h16)c[vg];
      } else {
#pragma unroll
        for (int vg = 0; vg < 8; vg++) VT[(ct * 16 + ln) * LDV + rt * 16 + vg + hi * 8] = (h16)c[vg];
      }
    }
    __syncthreads();
    // attention: wave w owns batches w and w+8
    for (int bi = 0; bi < 2; bi++) {
      int b = w + bi * 8, row0 = b * Lp;
      v8f s = vzero8();
      s = wmma_f16(ldsA(QB, LDA, row0, 0),  ldsB(KB, LDA, row0, 0),  s);
      s = wmma_f16(ldsA(QB, LDA, row0, 32), ldsB(KB, LDA, row0, 32), s);
      float* scb = SC + b * 256;
#pragma unroll
      for (int vg = 0; vg < 8; vg++) scb[(vg + hi * 8) * 16 + ln] = s[vg] * 0.125f;
      __syncthreads();
      // reference softmax is over the QUERY axis (axis=2) -> column-wise over q
      if (lane < 7) {
        float vb[7], m = -1e30f, sum = 0.f;
        for (int r = 0; r < 7; r++) { vb[r] = scb[r * 16 + lane]; m = fmaxf(m, vb[r]); }
        for (int r = 0; r < 7; r++) { vb[r] = __expf(vb[r] - m); sum += vb[r]; }
        float inv = 1.f / sum;
        for (int r = 0; r < 7; r++) AT[b * 512 + r * 32 + lane] = (h16)(vb[r] * inv);
      }
      __syncthreads();
      // O = A * V  (A zero-padded beyond 7x7, so garbage V rows vanish)
      for (int ct = 0; ct < 4; ct++) {
        v8f o = vzero8();
        o = wmma_f16(ldsA(AT + b * 512, 32, 0, 0), ldsB(VT, LDV, ct * 16, row0), o);
        if (hi == 0) {
#pragma unroll
          for (int vg = 0; vg < 8; vg++) XB[(row0 + vg) * LDA + ct * 16 + ln] = (h16)o[vg];
        }
      }
      __syncthreads();
    }
    // OC += O_h @ Wo_h
    if (w == 0) {
      tdm_load(OFF_WS, e_wo + h * 64 * 64, 64, 64, 64);
      __builtin_amdgcn_s_wait_tensorcnt(0);
    }
    __syncthreads();
    cvtWT(W0, LDA, WSf, 64, 64);
    __syncthreads();
    for (int j = w; j < 32; j += 8) {
      int rt = j >> 2, ct = j & 3;
      v8f c;
#pragma unroll
      for (int vg = 0; vg < 8; vg++) c[vg] = OC[(rt * 16 + vg + hi * 8) * LDO + ct * 16 + ln];
      c = wmma_f16(ldsA(XB, LDA, rt * 16, 0),  ldsB(W0, LDA, ct * 16, 0),  c);
      c = wmma_f16(ldsA(XB, LDA, rt * 16, 32), ldsB(W0, LDA, ct * 16, 32), c);
#pragma unroll
      for (int vg = 0; vg < 8; vg++) OC[(rt * 16 + vg + hi * 8) * LDO + ct * 16 + ln] = c[vg];
    }
    __syncthreads();
  }

  // ---- S3: + e_bo, LayerNorm1 -> XA ----
  if (tid < R) {
    int r = tid, l = r & 7;
    if (l < 7) {
      float sum = 0.f, sq = 0.f;
      for (int d = 0; d < 64; d++) { float x = OC[r * LDO + d] + e_bo[d]; sum += x; sq += x * x; }
      float mean = sum * (1.f / 64.f);
      float inv = rsqrtf(fmaxf(sq * (1.f / 64.f) - mean * mean, 0.f) + 1e-5f);
      for (int d = 0; d < 64; d++) {
        float x = OC[r * LDO + d] + e_bo[d];
        XA[r * LDA + d] = (h16)((x - mean) * inv * e_g1[d] + e_b1[d]);
      }
    } else {
      for (int d = 0; d < 64; d++) XA[r * LDA + d] = (h16)0.f;
    }
  }
  __syncthreads();

  // ---- S4: FFN 64->128->64 in two 64-wide halves, accumulate into OC ----
  for (int i = tid; i < ROWQ * LDO; i += 256) OC[i] = 0.f;
  __syncthreads();
  for (int hf = 0; hf < 2; hf++) {
    if (w == 0) {
      tdm_load(OFF_WS,         e_c1w + hf * 64,      128, 64, 64);  // c1w col window
      tdm_load(OFF_WS + 16384, e_c2w + hf * 64 * 64,  64, 64, 64);  // c2w row window
      __builtin_amdgcn_s_wait_tensorcnt(0);
    }
    __syncthreads();
    cvtWT(W0, LDA, WSf,        64, 64);
    cvtWT(W1, LDA, WSf + 4096, 64, 64);
    __syncthreads();
    for (int j = w; j < 32; j += 8) {   // H1 = relu(XA@W1half + b1half) -> XB
      int rt = j >> 2, ct = j & 3;
      v8f c = vzero8();
      c = wmma_f16(ldsA(XA, LDA, rt * 16, 0),  ldsB(W0, LDA, ct * 16, 0),  c);
      c = wmma_f16(ldsA(XA, LDA, rt * 16, 32), ldsB(W0, LDA, ct * 16, 32), c);
      int col = ct * 16 + ln;
      float bias = e_c1b[hf * 64 + col];
#pragma unroll
      for (int vg = 0; vg < 8; vg++)
        XB[(rt * 16 + vg + hi * 8) * LDA + col] = (h16)fmaxf(c[vg] + bias, 0.f);
    }
    __syncthreads();
    for (int j = w; j < 32; j += 8) {   // OC += XB @ W2half
      int rt = j >> 2, ct = j & 3;
      v8f c;
#pragma unroll
      for (int vg = 0; vg < 8; vg++) c[vg] = OC[(rt * 16 + vg + hi * 8) * LDO + ct * 16 + ln];
      c = wmma_f16(ldsA(XB, LDA, rt * 16, 0),  ldsB(W1, LDA, ct * 16, 0),  c);
      c = wmma_f16(ldsA(XB, LDA, rt * 16, 32), ldsB(W1, LDA, ct * 16, 32), c);
#pragma unroll
      for (int vg = 0; vg < 8; vg++) OC[(rt * 16 + vg + hi * 8) * LDO + ct * 16 + ln] = c[vg];
    }
    __syncthreads();
  }

  // ---- S5: x1 = relu(OC + c2b); x = LN2(x + x1) -> XA ----
  if (tid < R) {
    int r = tid, l = r & 7;
    if (l < 7) {
      float sum = 0.f, sq = 0.f;
      for (int d = 0; d < 64; d++) {
        float v = fmaxf(OC[r * LDO + d] + e_c2b[d], 0.f) + (float)XA[r * LDA + d];
        sum += v; sq += v * v;
      }
      float mean = sum * (1.f / 64.f);
      float inv = rsqrtf(fmaxf(sq * (1.f / 64.f) - mean * mean, 0.f) + 1e-5f);
      for (int d = 0; d < 64; d++) {
        float v = fmaxf(OC[r * LDO + d] + e_c2b[d], 0.f) + (float)XA[r * LDA + d];
        XA[r * LDA + d] = (h16)((v - mean) * inv * e_g2[d] + e_b2[d]);
      }
    } else {
      for (int d = 0; d < 64; d++) XA[r * LDA + d] = (h16)0.f;
    }
  }
  __syncthreads();

  // ---- S6: decoder. softmax over q-axis with 1 query => weights == 1, o_h = sum_k V_h[k] ----
  for (int h = 0; h < Hn; h++) {
    if (w == 0) {
      tdm_load(OFF_WS, d_wv + h * 4096, 64, 64, 64);
      __builtin_amdgcn_s_wait_tensorcnt(0);
    }
    __syncthreads();
    cvtWT(W0, LDA, WSf, 64, 64);
    __syncthreads();
    for (int j = w; j < 32; j += 8) {    // V_h = XA @ d_wv[h] -> QB (row major)
      int rt = j >> 2, ct = j & 3;
      v8f c = vzero8();
      c = wmma_f16(ldsA(XA, LDA, rt * 16, 0),  ldsB(W0, LDA, ct * 16, 0),  c);
      c = wmma_f16(ldsA(XA, LDA, rt * 16, 32), ldsB(W0, LDA, ct * 16, 32), c);
#pragma unroll
      for (int vg = 0; vg < 8; vg++) QB[(rt * 16 + vg + hi * 8) * LDA + ct * 16 + ln] = (h16)c[vg];
    }
    __syncthreads();
    for (int bi = 0; bi < 2; bi++) {
      int b = w + bi * 8, row0 = b * Lp;
#pragma unroll
      for (int half = 0; half < 2; half++) {
        int e = lane + half * 32;
        float o2 = 0.f;
        for (int j2 = 0; j2 < 7; j2++) o2 += (float)QB[(row0 + j2) * LDA + e];
        ODC[b * LDD + h * 64 + e] = (h16)o2;
      }
    }
    __syncthreads();
  }
  // o_cat(16x256) @ d_wo(256x64) + d_bo  -> OE (in AT region, ld=LDA)
  v8f oc2 = vzero8();
  for (int kc = 0; kc < 4; kc++) {
    if (w == 0) {
      tdm_load(OFF_WS, d_wo + kc * 64 * 64, 64, 64, 64);
      __builtin_amdgcn_s_wait_tensorcnt(0);
    }
    __syncthreads();
    cvtWT(W0, LDA, WSf, 64, 64);
    __syncthreads();
    if (w < 4) {
      oc2 = wmma_f16(ldsA(ODC, LDD, 0, kc * 64),      ldsB(W0, LDA, w * 16, 0),  oc2);
      oc2 = wmma_f16(ldsA(ODC, LDD, 0, kc * 64 + 32), ldsB(W0, LDA, w * 16, 32), oc2);
    }
    __syncthreads();
  }
  if (w < 4) {
    int col = w * 16 + ln;
    float bias = d_bo[col];
#pragma unroll
    for (int vg = 0; vg < 8; vg++) AT[(vg + hi * 8) * LDA + col] = (h16)(oc2[vg] + bias);
  }
  __syncthreads();
  // final fc: OE(16x64) @ fc_w(64x11) + fc_b
  for (int i = tid; i < 16 * LDA; i += 256) W0[i] = (h16)0.f;  // zero cols 11..15
  if (w == 0) {
    tdm_load(OFF_WS, fc_w, 11, 11, 64);
    __builtin_amdgcn_s_wait_tensorcnt(0);
  }
  __syncthreads();
  cvtWT(W0, LDA, WSf, 11, 64);
  __syncthreads();
  if (w == 0) {
    v8f c = vzero8();
    c = wmma_f16(ldsA(AT, LDA, 0, 0),  ldsB(W0, LDA, 0, 0),  c);
    c = wmma_f16(ldsA(AT, LDA, 0, 32), ldsB(W0, LDA, 0, 32), c);
    if (ln < 11) {
      float bias = fc_b[ln];
#pragma unroll
      for (int vg = 0; vg < 8; vg++) {
        int m2 = vg + hi * 8;
        out[(b0 + m2) * 11 + ln] = c[vg] + bias;
      }
    }
  }
}

extern "C" void kernel_launch(void* const* d_in, const int* in_sizes, int n_in,
                              void* d_out, int out_size, void* d_ws, size_t ws_size,
                              hipStream_t stream) {
  (void)in_sizes; (void)n_in; (void)out_size; (void)d_ws; (void)ws_size;
  const int*   tokens = (const int*)d_in[0];
  const float* emb    = (const float*)d_in[1];
  const float* pos    = (const float*)d_in[2];
  const float* qparam = (const float*)d_in[3];
  const float* in_w   = (const float*)d_in[4];
  const float* in_b   = (const float*)d_in[5];
  const float* e_wq   = (const float*)d_in[6];
  const float* e_wk   = (const float*)d_in[7];
  const float* e_wv   = (const float*)d_in[8];
  const float* e_wo   = (const float*)d_in[9];
  const float* e_bo   = (const float*)d_in[10];
  const float* e_g1   = (const float*)d_in[11];
  const float* e_b1   = (const float*)d_in[12];
  const float* e_c1w  = (const float*)d_in[13];
  const float* e_c1b  = (const float*)d_in[14];
  const float* e_c2w  = (const float*)d_in[15];
  const float* e_c2b  = (const float*)d_in[16];
  const float* e_g2   = (const float*)d_in[17];
  const float* e_b2   = (const float*)d_in[18];
  const float* d_wq   = (const float*)d_in[19];
  const float* d_wk   = (const float*)d_in[20];
  const float* d_wv   = (const float*)d_in[21];
  const float* d_wo   = (const float*)d_in[22];
  const float* d_bo   = (const float*)d_in[23];
  const float* fc_w   = (const float*)d_in[24];
  const float* fc_b   = (const float*)d_in[25];
  float* out = (float*)d_out;

  dim3 grid(32768 / Bt), block(256);
  xformer_mega<<<grid, block, SMEM, stream>>>(
      tokens, emb, pos, qparam, in_w, in_b, e_wq, e_wk, e_wv, e_wo, e_bo,
      e_g1, e_b1, e_c1w, e_c1b, e_c2w, e_c2b, e_g2, e_b2,
      d_wq, d_wk, d_wv, d_wo, d_bo, fc_w, fc_b, out);
}
